// Seq2Seq_12128987644395
// MI455X (gfx1250) — compile-verified, matrix operand order checked by reading, not run on hardware
//
#include <hip/hip_runtime.h>
#include <hip/hip_bf16.h>

// ---------------------------------------------------------------------------
// Seq2Seq GRU + attention + vocab projection for MI455X (gfx1250, wave32).
//  * logits GEMM (2528x32000x1024, 166 GFLOP) : bf16 WMMA, 64x256 block tile,
//    double-buffered LDS staging via async load-to-LDS (s_wait_asynccnt).
//  * GRU scans: persistent 8-block kernels, Whh stream spread over 8 WGPs,
//    per-step global atomic barrier; decoder attention runs in a 9th block
//    overlapped with the next step's recurrence GEMM.
// ---------------------------------------------------------------------------

typedef __attribute__((ext_vector_type(16))) __bf16 v16bf;
typedef __attribute__((ext_vector_type(8)))  __bf16 v8bf;
typedef __attribute__((ext_vector_type(8)))  float  v8f;
typedef __attribute__((ext_vector_type(4)))  int    v4i;

#define Vv 32000
#define Ee 256
#define Hh 512
#define Bb 32
#define Ss 80
#define Tt 80
#define NBLK 8              // GEMM blocks participating in the scan barrier
#define KC 64               // K-chunk staged to LDS in the logits GEMM

#define WMMA_BF16(a, b, c) \
    __builtin_amdgcn_wmma_f32_16x16x32_bf16(false, (a), false, (b), (short)0, (c), false, false)

#if __has_builtin(__builtin_amdgcn_global_load_async_to_lds_b128) && \
    __has_builtin(__builtin_amdgcn_s_wait_asynccnt)
#define USE_ASYNC_LDS 1
#else
#define USE_ASYNC_LDS 0
#endif

static __device__ __forceinline__ float sigm(float x) {
    return 1.0f / (1.0f + __expf(-x));
}

// Load a 16x32 bf16 tile (row-major, leading dim ld) in WMMA A/B layout:
// lane half = lane>>4, row = lane&15; two 16B chunks at K offsets 8*half and
// 16+8*half (CDNA5 ISA, "16-bit A-Matrix 16x32").
static __device__ __forceinline__ v16bf load_tile(const __bf16* base, int ld,
                                                  int row0, int k0, int lane) {
    int half = lane >> 4;
    int r    = lane & 15;
    const __bf16* p = base + (size_t)(row0 + r) * (size_t)ld + (k0 + 8 * half);
    v8bf lo = *(const v8bf*)(p);
    v8bf hi = *(const v8bf*)(p + 16);
    v16bf out;
#pragma unroll
    for (int i = 0; i < 8; ++i) { out[i] = lo[i]; out[8 + i] = hi[i]; }
    return out;
}

// 16-byte global -> LDS staging element (async on CDNA5, sync fallback).
static __device__ __forceinline__ void stage16(const __bf16* g, __bf16* l) {
#if USE_ASYNC_LDS
    __builtin_amdgcn_global_load_async_to_lds_b128(
        (__attribute__((address_space(1))) v4i*)(__bf16*)g,
        (__attribute__((address_space(3))) v4i*)l, 0, 0);
#else
    *(v8bf*)l = *(const v8bf*)g;
#endif
}
static __device__ __forceinline__ void stage_wait() {
#if USE_ASYNC_LDS
    __builtin_amdgcn_s_wait_asynccnt(0);
#endif
}

// ------------------------------- prologue ----------------------------------

__global__ void cvt_bf16_kernel(const float* __restrict__ src,
                                __bf16* __restrict__ dst, int n) {
    int i = blockIdx.x * blockDim.x + threadIdx.x;
    if (i < n) dst[i] = (__bf16)src[i];
}

// bias[j] = bih[j] + bhh[j] for r,z slices (j<1024); bih[j] only for n slice.
__global__ void fold_bias_kernel(const float* __restrict__ bih,
                                 const float* __restrict__ bhh,
                                 float* __restrict__ out) {
    int i = blockIdx.x * blockDim.x + threadIdx.x;
    if (i < 3 * Hh) out[i] = bih[i] + (i < 2 * Hh ? bhh[i] : 0.0f);
}

__global__ void gather_embed_kernel(const int* __restrict__ tok, int tok_ld,
                                    int cols, const float* __restrict__ emb,
                                    __bf16* __restrict__ out, int total) {
    int i = blockIdx.x * blockDim.x + threadIdx.x;
    if (i >= total) return;
    int r = i >> 8, e = i & 255;
    int b = r / cols, s = r % cols;
    int t = tok[b * tok_ld + s];
    out[i] = (__bf16)emb[(size_t)t * Ee + e];
}

__global__ void zero_t0_kernel(float* __restrict__ out) {
    int i = blockIdx.x * blockDim.x + threadIdx.x;
    if (i >= Bb * Vv) return;
    int b = i / Vv, v = i % Vv;
    out[(size_t)b * Tt * Vv + v] = 0.0f;
}

// Init h state (parity 0), reset scan barrier counter, zero A-pad rows.
__global__ void scan_init_kernel(float* __restrict__ hf, __bf16* __restrict__ hb,
                                 const float* __restrict__ h0_src,  // null -> 0
                                 unsigned* __restrict__ ctr,
                                 __bf16* __restrict__ pad, int pad_n) {
    int i = blockIdx.x * blockDim.x + threadIdx.x;
    if (i == 0) *ctr = 0u;
    if (i < Bb * Hh) {
        int b = i >> 9, j = i & (Hh - 1);
        float v = h0_src ? h0_src[((size_t)b * Ss + (Ss - 1)) * Hh + j] : 0.0f;
        hf[i] = v;
        hb[i] = (__bf16)v;
    }
    if (pad && i < pad_n) pad[i] = (__bf16)0.0f;
}

// ----------------------- generic bf16 WMMA GEMM ----------------------------
// C[M,N](f32) = A[M,K](bf16) @ B[N,K](bf16)^T + bias[N]; one wave per tile.

__global__ __launch_bounds__(256) void gemm_bf16_bias_kernel(
    const __bf16* __restrict__ A, const __bf16* __restrict__ B,
    const float* __restrict__ bias, float* __restrict__ C,
    int M, int N, int K) {
    int wave = threadIdx.x >> 5, lane = threadIdx.x & 31;
    int ntiles = N >> 4;
    int tile = blockIdx.x * 8 + wave;
    if (tile >= (M >> 4) * ntiles) return;
    int mt = tile / ntiles, nt = tile % ntiles;
    v8f acc = {};
    for (int k = 0; k < K; k += 32) {
        v16bf a = load_tile(A, K, mt * 16, k, lane);
        v16bf b = load_tile(B, K, nt * 16, k, lane);
        acc = WMMA_BF16(a, b, acc);
    }
    int half = lane >> 4, ln = lane & 15;
    int col = nt * 16 + ln;
    float bs = bias ? bias[col] : 0.0f;
#pragma unroll
    for (int g = 0; g < 8; ++g)
        C[(size_t)(mt * 16 + g + 8 * half) * N + col] = acc[g] + bs;
}

// ----------------------- persistent GRU scan kernel ------------------------
// Blocks 0..7: per step, stage h(bf16) to LDS, gh = h@Whh^T via WMMA (each
// wave owns one (m-tile, j-tile) pair and all 3 gate accumulators), gates
// fused in epilogue, h ping-pongs in global (f32 carry + bf16 operand).
// Per-step sync: global atomic arrive + spin (8 arrivals per step).
// Block 8 (decoder only): attention for step t, overlapped with step t+1.

__global__ __launch_bounds__(256) void gru_scan_kernel(
    const float* __restrict__ gi,        // [B*t_rows, 1536], biases folded
    const __bf16* __restrict__ Whh,      // [1536, 512] bf16
    const float* __restrict__ bhh,       // [1536] f32 (n-slice used)
    float* __restrict__ hf,              // [2][B*H] f32 carry (ping-pong)
    __bf16* __restrict__ hb,             // [2][B*H] bf16 operand (ping-pong)
    float* __restrict__ enc_store,       // [B,S,H] (encoder) or null
    const float* __restrict__ enc_read,  // [B,S,H] (decoder attn) or null
    __bf16* __restrict__ combined,       // [(T-1)*B, 2H] (decoder) or null
    unsigned* __restrict__ ctr, int steps, int t_rows) {
    __shared__ __bf16 hs[Bb * Hh];   // 32 KB staged h
    __shared__ float  sm[Bb * Ss];   // attention scores (block 8 only)

    int tid = threadIdx.x, blk = blockIdx.x;
    int lane = tid & 31, wave = tid >> 5;
    int half = lane >> 4, ln = lane & 15;

    if (blk < NBLK) {
        int task = blk * 8 + wave;       // 0..63
        int m    = task & 1;             // m-tile (batch rows m*16..m*16+15)
        int jt   = task >> 1;            // j-tile 0..31
        int j    = jt * 16 + ln;
        float bhn = bhh[2 * Hh + j];
        for (int t = 0; t < steps; ++t) {
            int p = t & 1;
            // stage h(bf16, parity p) into LDS (shared A operand)
            for (int i = tid * 8; i < Bb * Hh; i += 256 * 8)
                *(v8bf*)&hs[i] = *(const v8bf*)&hb[p * Bb * Hh + i];
            __syncthreads();

            v8f ar = {}, az = {}, an = {};
            for (int k = 0; k < Hh; k += 32) {
                v16bf a  = load_tile(hs, Hh, m * 16, k, lane);
                v16bf br = load_tile(Whh, Hh, 0 * Hh + jt * 16, k, lane);
                v16bf bz = load_tile(Whh, Hh, 1 * Hh + jt * 16, k, lane);
                v16bf bn = load_tile(Whh, Hh, 2 * Hh + jt * 16, k, lane);
                ar = WMMA_BF16(a, br, ar);
                az = WMMA_BF16(a, bz, az);
                an = WMMA_BF16(a, bn, an);
            }
#pragma unroll
            for (int g = 0; g < 8; ++g) {
                int b = m * 16 + g + 8 * half;
                const float* gir = gi + ((size_t)b * t_rows + t) * (3 * Hh);
                float r = sigm(gir[j] + ar[g]);
                float z = sigm(gir[Hh + j] + az[g]);
                float n = tanhf(gir[2 * Hh + j] + r * (an[g] + bhn));
                float hold = hf[p * Bb * Hh + b * Hh + j];
                float hnew = (1.0f - z) * n + z * hold;
                hf[(p ^ 1) * Bb * Hh + b * Hh + j] = hnew;
                hb[(p ^ 1) * Bb * Hh + b * Hh + j] = (__bf16)hnew;
                if (enc_store)
                    enc_store[((size_t)b * Ss + t) * Hh + j] = hnew;
            }
            __threadfence();
            __syncthreads();
            if (tid == 0) {
                __hip_atomic_fetch_add(ctr, 1u, __ATOMIC_RELEASE,
                                       __HIP_MEMORY_SCOPE_AGENT);
                unsigned tgt = (unsigned)NBLK * (unsigned)(t + 1);
                while (__hip_atomic_load(ctr, __ATOMIC_ACQUIRE,
                                         __HIP_MEMORY_SCOPE_AGENT) < tgt)
                    __builtin_amdgcn_s_sleep(1);
            }
            __syncthreads();
            __threadfence();
        }
    } else {
        // ---- attention block (decoder): consumes h_t, never arrives ----
        for (int t = 0; t < steps; ++t) {
            if (tid == 0) {
                unsigned tgt = (unsigned)NBLK * (unsigned)(t + 1);
                while (__hip_atomic_load(ctr, __ATOMIC_ACQUIRE,
                                         __HIP_MEMORY_SCOPE_AGENT) < tgt)
                    __builtin_amdgcn_s_sleep(1);
            }
            __syncthreads();
            __threadfence();
            const float* hcur = hf + ((t & 1) ^ 1) * Bb * Hh;   // h_t
            // scores[b,s] = enc_outs[b,s,:] . h
            for (int idx = tid; idx < Bb * Ss; idx += 256) {
                int b = idx / Ss, s = idx % Ss;
                const float* eo = enc_read + ((size_t)b * Ss + s) * Hh;
                const float* hv = hcur + b * Hh;
                float d = 0.0f;
                for (int jj = 0; jj < Hh; ++jj) d += eo[jj] * hv[jj];
                sm[idx] = d;
            }
            __syncthreads();
            if (tid < Bb) {     // softmax, one thread per batch row
                float mx = -1e30f;
                for (int s = 0; s < Ss; ++s) mx = fmaxf(mx, sm[tid * Ss + s]);
                float sum = 0.0f;
                for (int s = 0; s < Ss; ++s) {
                    float e = __expf(sm[tid * Ss + s] - mx);
                    sm[tid * Ss + s] = e;
                    sum += e;
                }
                float inv = 1.0f / sum;
                for (int s = 0; s < Ss; ++s) sm[tid * Ss + s] *= inv;
            }
            __syncthreads();
            for (int idx = tid; idx < Bb * Hh; idx += 256) {
                int b = idx >> 9, jj = idx & (Hh - 1);
                float c = 0.0f;
                for (int s = 0; s < Ss; ++s)
                    c += sm[b * Ss + s] * enc_read[((size_t)b * Ss + s) * Hh + jj];
                size_t row = (size_t)t * Bb + b;
                combined[row * (2 * Hh) + jj]      = (__bf16)hcur[b * Hh + jj];
                combined[row * (2 * Hh) + Hh + jj] = (__bf16)c;
            }
            __syncthreads();
        }
    }
}

// ------------------------ vocab projection GEMM ----------------------------
// logits[b, t+1, :] = combined[t*32+b, :] @ out_W^T + out_b.
// Block tile 64M x 256N, K chunked by 64 with double-buffered LDS staging
// (async load-to-LDS on CDNA5). Wave (mp,nq): 2 m-tiles x 4 n-tiles.

__global__ __launch_bounds__(256) void logits_gemm_kernel(
    const __bf16* __restrict__ A,        // [2560(pad), 1024]
    const __bf16* __restrict__ W,        // [32000, 1024]
    const float* __restrict__ out_b,     // [32000]
    float* __restrict__ out) {           // [B, T, V]
    __shared__ __bf16 As[2][64 * KC];    // 2 x 8 KB
    __shared__ __bf16 Bs[2][256 * KC];   // 2 x 32 KB

    int tid = threadIdx.x;
    int lane = tid & 31, wave = tid >> 5;
    int half = lane >> 4, ln = lane & 15;
    int mb = blockIdx.x / 125, nb = blockIdx.x % 125;
    int m0 = mb * 64, n0 = nb * 256;
    int mp = wave >> 2, nq = wave & 3;

    v8f acc[2][4];
#pragma unroll
    for (int i = 0; i < 2; ++i)
#pragma unroll
        for (int q = 0; q < 4; ++q) acc[i][q] = (v8f){};

    // stage one K-chunk (64 cols): A 64 rows + W 256 rows, 2560 x 16B segs
    auto stage_chunk = [&](int buf, int kc) {
#pragma unroll
        for (int it = 0; it < 10; ++it) {
            int s = tid + it * 256;
            if (s < 512) {
                int row = s >> 3, off = (s & 7) * 8;
                stage16(A + (size_t)(m0 + row) * 1024 + kc + off,
                        &As[buf][row * KC + off]);
            } else {
                int s2 = s - 512;
                int row = s2 >> 3, off = (s2 & 7) * 8;
                stage16(W + (size_t)(n0 + row) * 1024 + kc + off,
                        &Bs[buf][row * KC + off]);
            }
        }
    };

    stage_chunk(0, 0);
    for (int c = 0; c < 1024 / KC; ++c) {
        int buf = c & 1;
        stage_wait();
        __syncthreads();                         // chunk c resident in LDS
        if (c + 1 < 1024 / KC) stage_chunk(buf ^ 1, (c + 1) * KC);
#pragma unroll
        for (int ks = 0; ks < 2; ++ks) {
            v16bf a0 = load_tile(As[buf], KC, (2 * mp + 0) * 16, ks * 32, lane);
            v16bf a1 = load_tile(As[buf], KC, (2 * mp + 1) * 16, ks * 32, lane);
#pragma unroll
            for (int q = 0; q < 4; ++q) {
                v16bf b = load_tile(Bs[buf], KC, (4 * nq + q) * 16, ks * 32, lane);
                acc[0][q] = WMMA_BF16(a0, b, acc[0][q]);
                acc[1][q] = WMMA_BF16(a1, b, acc[1][q]);
            }
        }
        __syncthreads();                         // done reading buf
    }

#pragma unroll
    for (int q = 0; q < 4; ++q) {
        int col = n0 + (4 * nq + q) * 16 + ln;
        float bs = out_b[col];
#pragma unroll
        for (int i = 0; i < 2; ++i) {
#pragma unroll
            for (int g = 0; g < 8; ++g) {
                int r = m0 + (2 * mp + i) * 16 + g + 8 * half;  // r = t*32 + b
                if (r < (Tt - 1) * Bb) {
                    int b = r & (Bb - 1), t = r >> 5;
                    out[(size_t)b * Tt * Vv + (size_t)(t + 1) * Vv + col] =
                        acc[i][q][g] + bs;
                }
            }
        }
    }
}

// ------------------------------- launcher ----------------------------------

extern "C" void kernel_launch(void* const* d_in, const int* in_sizes, int n_in,
                              void* d_out, int out_size, void* d_ws, size_t ws_size,
                              hipStream_t stream) {
    const int*   src     = (const int*)d_in[0];
    const int*   tgt     = (const int*)d_in[1];
    const float* emb     = (const float*)d_in[2];
    const float* enc_Wih = (const float*)d_in[3];
    const float* enc_Whh = (const float*)d_in[4];
    const float* enc_bih = (const float*)d_in[5];
    const float* enc_bhh = (const float*)d_in[6];
    const float* dec_Wih = (const float*)d_in[7];
    const float* dec_Whh = (const float*)d_in[8];
    const float* dec_bih = (const float*)d_in[9];
    const float* dec_bhh = (const float*)d_in[10];
    const float* out_W   = (const float*)d_in[11];
    const float* out_b   = (const float*)d_in[12];
    float* out = (float*)d_out;

    char* w = (char*)d_ws;
    size_t off = 0;
    auto alloc = [&](size_t bytes) -> void* {
        void* p = w + off;
        off += (bytes + 255) & ~(size_t)255;
        return p;
    };

    const int M_enc = Bb * Ss;          // 2560
    const int M_dec = Bb * (Tt - 1);    // 2528
    const int M_pad = 2560;             // logits GEMM A rows (64-aligned)

    __bf16*   xsrc_bf  = (__bf16*)alloc((size_t)M_enc * Ee * 2);
    __bf16*   xdec_bf  = (__bf16*)alloc((size_t)M_dec * Ee * 2);
    __bf16*   eWih_bf  = (__bf16*)alloc((size_t)3 * Hh * Ee * 2);
    __bf16*   dWih_bf  = (__bf16*)alloc((size_t)3 * Hh * Ee * 2);
    __bf16*   eWhh_bf  = (__bf16*)alloc((size_t)3 * Hh * Hh * 2);
    __bf16*   dWhh_bf  = (__bf16*)alloc((size_t)3 * Hh * Hh * 2);
    __bf16*   outW_bf  = (__bf16*)alloc((size_t)Vv * 2 * Hh * 2);
    float*    bias_enc = (float*)alloc((size_t)3 * Hh * 4);
    float*    bias_dec = (float*)alloc((size_t)3 * Hh * 4);
    float*    gi_enc   = (float*)alloc((size_t)M_enc * 3 * Hh * 4);
    float*    gi_dec   = (float*)alloc((size_t)M_dec * 3 * Hh * 4);
    float*    enc_outs = (float*)alloc((size_t)Bb * Ss * Hh * 4);
    __bf16*   combined = (__bf16*)alloc((size_t)M_pad * 2 * Hh * 2);
    float*    h_f      = (float*)alloc((size_t)2 * Bb * Hh * 4);
    __bf16*   h_b      = (__bf16*)alloc((size_t)2 * Bb * Hh * 2);
    unsigned* ctr_enc  = (unsigned*)alloc(256);
    unsigned* ctr_dec  = (unsigned*)alloc(256);

    // ---- prologue ----
    auto cvt = [&](const float* s, __bf16* d, int n) {
        cvt_bf16_kernel<<<(n + 255) / 256, 256, 0, stream>>>(s, d, n);
    };
    cvt(enc_Wih, eWih_bf, 3 * Hh * Ee);
    cvt(dec_Wih, dWih_bf, 3 * Hh * Ee);
    cvt(enc_Whh, eWhh_bf, 3 * Hh * Hh);
    cvt(dec_Whh, dWhh_bf, 3 * Hh * Hh);
    cvt(out_W,   outW_bf, Vv * 2 * Hh);

    fold_bias_kernel<<<6, 256, 0, stream>>>(enc_bih, enc_bhh, bias_enc);
    fold_bias_kernel<<<6, 256, 0, stream>>>(dec_bih, dec_bhh, bias_dec);

    gather_embed_kernel<<<(M_enc * Ee + 255) / 256, 256, 0, stream>>>(
        src, Ss, Ss, emb, xsrc_bf, M_enc * Ee);
    gather_embed_kernel<<<(M_dec * Ee + 255) / 256, 256, 0, stream>>>(
        tgt, Tt, Tt - 1, emb, xdec_bf, M_dec * Ee);

    // ---- time-parallel input GEMMs: gi = x @ Wih^T + folded bias ----
    {
        int tiles = (M_enc / 16) * (3 * Hh / 16);
        gemm_bf16_bias_kernel<<<(tiles + 7) / 8, 256, 0, stream>>>(
            xsrc_bf, eWih_bf, bias_enc, gi_enc, M_enc, 3 * Hh, Ee);
    }
    {
        int tiles = (M_dec / 16) * (3 * Hh / 16);
        gemm_bf16_bias_kernel<<<(tiles + 7) / 8, 256, 0, stream>>>(
            xdec_bf, dWih_bf, bias_dec, gi_dec, M_dec, 3 * Hh, Ee);
    }

    // ---- encoder scan: init h=0 then 8-block persistent scan ----
    scan_init_kernel<<<128, 256, 0, stream>>>(h_f, h_b, nullptr, ctr_enc,
                                              nullptr, 0);
    gru_scan_kernel<<<NBLK, 256, 0, stream>>>(
        gi_enc, eWhh_bf, enc_bhh, h_f, h_b, enc_outs, nullptr, nullptr,
        ctr_enc, Ss, Ss);

    // ---- decoder scan: init h=h_enc, zero combined pad rows ----
    scan_init_kernel<<<128, 256, 0, stream>>>(
        h_f, h_b, enc_outs, ctr_dec,
        combined + (size_t)M_dec * 2 * Hh, (M_pad - M_dec) * 2 * Hh);
    gru_scan_kernel<<<NBLK + 1, 256, 0, stream>>>(
        gi_dec, dWhh_bf, dec_bhh, h_f, h_b, nullptr, enc_outs, combined,
        ctr_dec, Tt - 1, Tt - 1);

    // ---- logits ----
    zero_t0_kernel<<<(Bb * Vv + 255) / 256, 256, 0, stream>>>(out);
    logits_gemm_kernel<<<(M_pad / 64) * (Vv / 256), 256, 0, stream>>>(
        combined, outW_bf, out_b, out);

    (void)in_sizes; (void)n_in; (void)out_size; (void)ws_size;
}
